// LateInteraction_62002147885421
// MI455X (gfx1250) — matrix operand order repaired and checked by reading, not compiled, not run on hardware
//
#include <hip/hip_runtime.h>
#include <math.h>
#include <stdint.h>

typedef __attribute__((ext_vector_type(16))) _Float16 v16h;
typedef __attribute__((ext_vector_type(8)))  _Float16 v8h;
typedef __attribute__((ext_vector_type(8)))  float    v8f;
typedef __attribute__((address_space(3)))    _Float16 lds_h;

#define B_  32
#define S_  256
#define H_  768
#define KC  32          // K-chunk = one WMMA K-step
#define NCH (H_ / KC)   // 24 chunks
#define LDSTRIDE 40     // 32 halves + 8 pad (16B) to break bank conflicts

// ---------------- Kernel 1: L2 normalize + f32 -> f16 ----------------
__global__ __launch_bounds__(256) void li_normalize(
    const float* __restrict__ q, const float* __restrict__ k,
    _Float16* __restrict__ qh, _Float16* __restrict__ kh)
{
    const int row = blockIdx.x;            // 0 .. 2*B*S-1
    const int tid = threadIdx.x;           // 256 threads, 3 elems each (H=768)
    const float* src;
    _Float16* dst;
    if (row < B_ * S_) { src = q + (size_t)row * H_;              dst = qh + (size_t)row * H_; }
    else               { const int r2 = row - B_ * S_;
                         src = k + (size_t)r2 * H_;               dst = kh + (size_t)r2 * H_; }

    float v0 = src[tid], v1 = src[tid + 256], v2 = src[tid + 512];
    float ss = v0 * v0 + v1 * v1 + v2 * v2;
    #pragma unroll
    for (int off = 16; off >= 1; off >>= 1) ss += __shfl_xor(ss, off, 32);

    __shared__ float sred[8];
    if ((tid & 31) == 0) sred[tid >> 5] = ss;
    __syncthreads();
    float tot = sred[0] + sred[1] + sred[2] + sred[3]
              + sred[4] + sred[5] + sred[6] + sred[7];
    float inv = 1.0f / fmaxf(sqrtf(tot), 1e-12f);
    dst[tid]       = (_Float16)(v0 * inv);
    dst[tid + 256] = (_Float16)(v1 * inv);
    dst[tid + 512] = (_Float16)(v2 * inv);
}

// ---------------- Kernel 2: decay table exp(-softplus(alpha)*|s-t|) ----------------
__global__ __launch_bounds__(256) void li_decay(
    const float* __restrict__ alpha_raw, float* __restrict__ decay)
{
    const int s = blockIdx.x, t = threadIdx.x;
    float a = alpha_raw[0];
    float alpha = (a > 20.0f) ? a : log1pf(__expf(a));
    decay[s * S_ + t] = __expf(-alpha * fabsf((float)(s - t)));
}

// ---------------- Kernel 3: fused pair GEMM + softmax + score ----------------
// grid (j=32, i=32), 512 threads = 16 waves; wave w owns rows [16w, 16w+16).
// Async-DMA double-buffered K pipeline: chunk ch+1 streams into LDS while
// WMMAs consume chunk ch.
__global__ __launch_bounds__(512) void li_pair_kernel(
    const _Float16* __restrict__ qh, const _Float16* __restrict__ kh,
    const float* __restrict__ decay, const float* __restrict__ q_mask,
    const float* __restrict__ k_mask, const float* __restrict__ logit_scale,
    float* __restrict__ out)
{
    __shared__ _Float16 ksh[2][S_ * LDSTRIDE];   // 2 x 20,480 B
    __shared__ float red[16];

    const int j    = blockIdx.x;
    const int i    = blockIdx.y;
    const int tid  = threadIdx.x;
    const int wv   = tid >> 5;
    const int lane = tid & 31;
    const int ln   = lane & 15;          // N (or M-row) sub-index within tile
    const int hh   = lane >> 4;          // lane half (0: lanes 0-15, 1: 16-31)
    const int sr   = wv * 16;            // first s-row owned by this wave

    const float NEGINF = -__builtin_inff();

    v8f acc[16];
    const v8f vzero = {0.f, 0.f, 0.f, 0.f, 0.f, 0.f, 0.f, 0.f};
    #pragma unroll
    for (int tc = 0; tc < 16; ++tc) acc[tc] = vzero;

    const _Float16* qbase = qh + (size_t)(i * S_ + sr + ln) * H_;
    const _Float16* kbase = kh + (size_t)(j * S_) * H_;

    // --- prologue: async-issue chunk 0 into buffer 0, prefetch A frag 0 ---
    #pragma unroll
    for (int g0 = 0; g0 < 2; ++g0) {
        const int g   = tid + g0 * 512;          // 1024 groups of 8 halves
        const int row = g >> 2;
        const int c8  = (g & 3) * 8;
        unsigned lofs = (unsigned)(uintptr_t)(lds_h*)(&ksh[0][row * LDSTRIDE + c8]);
        unsigned long long ga =
            (unsigned long long)(uintptr_t)(kbase + (size_t)row * H_ + c8);
        asm volatile("global_load_async_to_lds_b128 %0, %1, off"
                     :: "v"(lofs), "v"(ga) : "memory");
    }
    v8h alo = *(const v8h*)(qbase + hh * 8);
    v8h ahi = *(const v8h*)(qbase + hh * 8 + 16);

    for (int ch = 0; ch < NCH; ++ch) {
        asm volatile("s_wait_asynccnt 0x0" ::: "memory");
        __syncthreads();                         // chunk ch resident for all waves

        // async-issue chunk ch+1 into the other buffer (overlaps WMMAs below)
        if (ch + 1 < NCH) {
            const int kb = (ch + 1) * KC;
            const _Float16* src = kbase + kb;
            _Float16* dstbuf = &ksh[(ch + 1) & 1][0];
            #pragma unroll
            for (int g0 = 0; g0 < 2; ++g0) {
                const int g   = tid + g0 * 512;
                const int row = g >> 2;
                const int c8  = (g & 3) * 8;
                unsigned lofs = (unsigned)(uintptr_t)(lds_h*)(&dstbuf[row * LDSTRIDE + c8]);
                unsigned long long ga =
                    (unsigned long long)(uintptr_t)(src + (size_t)row * H_ + c8);
                asm volatile("global_load_async_to_lds_b128 %0, %1, off"
                             :: "v"(lofs), "v"(ga) : "memory");
            }
        }

        // A fragment for this chunk (preloaded); prefetch next chunk's A
        v16h a = __builtin_shufflevector(alo, ahi,
                     0,1,2,3,4,5,6,7,8,9,10,11,12,13,14,15);
        if (ch + 1 < NCH) {
            const _Float16* ap = qbase + (ch + 1) * KC + hh * 8;
            alo = *(const v8h*)(ap);
            ahi = *(const v8h*)(ap + 16);
        }

        // 16 WMMAs on chunk ch, B fragments double-buffered from LDS
        const _Float16* kshbuf = &ksh[ch & 1][0];
        v8h blo = *(const v8h*)(kshbuf + ln * LDSTRIDE + hh * 8);
        v8h bhi = *(const v8h*)(kshbuf + ln * LDSTRIDE + hh * 8 + 16);
        #pragma unroll
        for (int tc = 0; tc < 16; ++tc) {
            v16h b = __builtin_shufflevector(blo, bhi,
                         0,1,2,3,4,5,6,7,8,9,10,11,12,13,14,15);
            if (tc < 15) {
                const _Float16* bp = kshbuf + ((tc + 1) * 16 + ln) * LDSTRIDE + hh * 8;
                blo = *(const v8h*)(bp);
                bhi = *(const v8h*)(bp + 16);
            }
            acc[tc] = __builtin_amdgcn_wmma_f32_16x16x32_f16(
                          false, a, false, b, (short)0, acc[tc], false, false);
        }
        __syncthreads();                         // close readers of buffer ch&1
    }

    // ---- epilogue: decayed masked softmax over t, score = sum p * sim_raw ----
    const float scale = __expf(logit_scale[0]);
    float partial = 0.0f;

    #pragma unroll
    for (int r = 0; r < 8; ++r) {
        const int s = sr + r + hh * 8;           // C layout: VGPR r -> M = r + 8*half
        const float qm = q_mask[i * S_ + s];
        const bool rowvalid = qm > 0.0f;
        const float* drow = decay + (size_t)s * S_;

        float l[16];
        float m = NEGINF;
        #pragma unroll
        for (int tc = 0; tc < 16; ++tc) {
            const int t = tc * 16 + ln;          // C layout: lane -> N
            const float km  = k_mask[j * S_ + t];
            const float sim = acc[tc][r];
            const float lg  = (rowvalid && km > 0.0f) ? scale * sim * drow[t] : NEGINF;
            l[tc] = lg;
            m = fmaxf(m, lg);
        }
        #pragma unroll
        for (int off = 8; off >= 1; off >>= 1) m = fmaxf(m, __shfl_xor(m, off, 32));

        float esum = 0.0f, num = 0.0f;
        if (m > NEGINF) {
            #pragma unroll
            for (int tc = 0; tc < 16; ++tc) {
                const float e = __expf(l[tc] - m);   // exp(-inf)=0 for masked t
                esum += e;
                num  += e * acc[tc][r];              // raw (undecayed) sim
            }
        }
        #pragma unroll
        for (int off = 8; off >= 1; off >>= 1) {
            esum += __shfl_xor(esum, off, 32);
            num  += __shfl_xor(num,  off, 32);
        }
        const float sc = (m > NEGINF && esum > 0.0f) ? (num / esum) * qm : 0.0f;
        if (ln == 0) partial += sc;               // one representative per half
    }

    #pragma unroll
    for (int off = 16; off >= 1; off >>= 1) partial += __shfl_xor(partial, off, 32);
    if (lane == 0) red[wv] = partial;
    __syncthreads();

    if (wv == 0) {
        float v = (lane < 16) ? red[lane] : 0.0f;
        #pragma unroll
        for (int off = 8; off >= 1; off >>= 1) v += __shfl_xor(v, off, 32);
        float dm = 0.0f;
        #pragma unroll
        for (int t2 = 0; t2 < 8; ++t2) dm += q_mask[i * S_ + lane + 32 * t2];
        #pragma unroll
        for (int off = 16; off >= 1; off >>= 1) dm += __shfl_xor(dm, off, 32);
        if (lane == 0) out[i * B_ + j] = v / fmaxf(dm, 1.0f);
    }
}

// ---------------- host side ----------------
extern "C" void kernel_launch(void* const* d_in, const int* in_sizes, int n_in,
                              void* d_out, int out_size, void* d_ws, size_t ws_size,
                              hipStream_t stream)
{
    (void)in_sizes; (void)n_in; (void)out_size; (void)ws_size;
    const float* q           = (const float*)d_in[0];
    const float* k           = (const float*)d_in[1];
    const float* qm          = (const float*)d_in[2];
    const float* km          = (const float*)d_in[3];
    const float* alpha_raw   = (const float*)d_in[4];
    const float* logit_scale = (const float*)d_in[5];
    float* out = (float*)d_out;

    char* ws = (char*)d_ws;
    const size_t embBytes = (size_t)B_ * S_ * H_ * sizeof(_Float16); // ~12.6 MB each
    _Float16* qh = (_Float16*)ws;
    _Float16* kh = (_Float16*)(ws + embBytes);
    float*  decay = (float*)(ws + 2 * embBytes);                     // 256 KB

    li_normalize<<<2 * B_ * S_, 256, 0, stream>>>(q, k, qh, kh);
    li_decay<<<S_, S_, 0, stream>>>(alpha_raw, decay);
    li_pair_kernel<<<dim3(B_, B_), 512, 0, stream>>>(qh, kh, decay, qm, km,
                                                     logit_scale, out);
}
